// MultiHeadAttention_84396107366683
// MI455X (gfx1250) — matrix operand order repaired and checked by reading, not compiled, not run on hardware
//
#include <hip/hip_runtime.h>

// Problem dims (fixed by the reference)
#define BB 8
#define TT 1024
#define DD 512
#define HH 8
#define KDIM 64
#define HK 512        // H*KD
#define LDS_STRIDE 33 // 16x32 f32 stage, odd stride to avoid LDS bank conflicts

typedef __attribute__((ext_vector_type(16))) __bf16 v16bf;
typedef __attribute__((ext_vector_type(8)))  __bf16 v8bf;
typedef __attribute__((ext_vector_type(8)))  float  v8f;

// D = A(16x32 bf16) * B(32x16 bf16) + C(16x16 f32)
#define WMMA_BF16(a, b, c) \
  __builtin_amdgcn_wmma_f32_16x16x32_bf16(false, (a), false, (b), (short)0, (c), false, false)

// ---- fragment loaders (contiguous k order: elem j of lane half h -> k = 16*h + j, same in A and B) ----

// A fragment from row-major bf16 (M x K tile, ld in elements).
__device__ __forceinline__ v16bf load_A_bf(const __bf16* base, long ld, int lane) {
  int half = (lane >> 4) & 1, m = lane & 15;
  const __bf16* row = base + (long)m * ld;
  v8bf lo = *(const v8bf*)(row + half * 16);
  v8bf hi = *(const v8bf*)(row + half * 16 + 8);
  v16bf a;
#pragma unroll
  for (int j = 0; j < 8; ++j) { a[j] = lo[j]; a[j + 8] = hi[j]; }
  return a;
}

// A fragment from row-major f32 source, convert to bf16.
__device__ __forceinline__ v16bf load_A_f32(const float* base, long ld, int lane) {
  int half = (lane >> 4) & 1, m = lane & 15;
  const float* row = base + (long)m * ld;
  v16bf a;
#pragma unroll
  for (int j = 0; j < 8; ++j) {
    a[j]     = (__bf16)row[half * 16 + j];
    a[j + 8] = (__bf16)row[half * 16 + 8 + j];
  }
  return a;
}

// A fragment from the per-wave LDS staging tile (f32, stride LDS_STRIDE).
__device__ __forceinline__ v16bf load_A_lds(const float* stage, int lane) {
  int half = (lane >> 4) & 1, m = lane & 15;
  const float* row = stage + m * LDS_STRIDE;
  v16bf a;
#pragma unroll
  for (int j = 0; j < 8; ++j) {
    a[j]     = (__bf16)row[half * 16 + j];
    a[j + 8] = (__bf16)row[half * 16 + 8 + j];
  }
  return a;
}

// B fragment where B(k,n) = src[n*ld + k]  (i.e. B = K^T, src row-major N x K).
// lane half h, col n: elem j -> K = j + 16*h  => one contiguous 32B load.
__device__ __forceinline__ v16bf load_B_nk_bf(const __bf16* base, long ld, int lane) {
  int half = (lane >> 4) & 1, n = lane & 15;
  return *(const v16bf*)(base + (long)n * ld + half * 16);
}

// B fragment from row-major K x N bf16 source (strided per element).
__device__ __forceinline__ v16bf load_B_kn_bf(const __bf16* base, long ld, int lane) {
  int half = (lane >> 4) & 1, n = lane & 15;
  v16bf b;
#pragma unroll
  for (int j = 0; j < 16; ++j) b[j] = base[(long)(j + 16 * half) * ld + n];
  return b;
}

// B fragment from row-major K x N f32 source, convert to bf16.
__device__ __forceinline__ v16bf load_B_kn_f32(const float* base, long ld, int lane) {
  int half = (lane >> 4) & 1, n = lane & 15;
  v16bf b;
#pragma unroll
  for (int j = 0; j < 16; ++j) b[j] = (__bf16)base[(long)(j + 16 * half) * ld + n];
  return b;
}

// Per-lane async copy of 16B global -> LDS (GLOBAL_LOAD_ASYNC_TO_LDS_B128,
// tracked by ASYNCcnt; low 32 bits of a generic LDS pointer == LDS offset).
__device__ __forceinline__ void async_copy16(const __bf16* gsrc, __bf16* lds_dst) {
  unsigned lo = (unsigned)(unsigned long long)lds_dst;
  asm volatile("global_load_async_to_lds_b128 %0, %1, off"
               :: "v"(lo), "v"(gsrc) : "memory");
}

// ---- kernels ----

// Y[8192x512](bf16) = X[8192x512](f32) @ W[512x512](f32).
// One 16x64 strip per wave: 1 A-frag feeds 4 WMMAs.
__global__ void __launch_bounds__(256) proj_kernel(const float* __restrict__ X,
                                                   const float* __restrict__ W,
                                                   __bf16* __restrict__ Y) {
  int lane = threadIdx.x & 31, wave = threadIdx.x >> 5;
  int tile = blockIdx.x * 8 + wave;      // 4096 strips = 512 * 8
  int tj = tile & 7;                     // 64-col block
  int tm = tile >> 3;                    // 16-row block (8 waves share tm)
  const float* A0 = X + (long)tm * 16 * DD;
  const float* B0 = W + tj * 64;
  v8f c0 = {}, c1 = {}, c2 = {}, c3 = {};
  for (int k0 = 0; k0 < DD; k0 += 32) {
    v16bf a = load_A_f32(A0 + k0, DD, lane);
    const float* Bk = B0 + (long)k0 * HK;
    c0 = WMMA_BF16(a, load_B_kn_f32(Bk +  0, HK, lane), c0);
    c1 = WMMA_BF16(a, load_B_kn_f32(Bk + 16, HK, lane), c1);
    c2 = WMMA_BF16(a, load_B_kn_f32(Bk + 32, HK, lane), c2);
    c3 = WMMA_BF16(a, load_B_kn_f32(Bk + 48, HK, lane), c3);
  }
  int half = lane >> 4, n = lane & 15;
  __bf16* Yp = Y + (long)tm * 16 * HK + tj * 64;
#pragma unroll
  for (int i = 0; i < 8; ++i) {
    long r = (long)(i + half * 8) * HK;
    Yp[r + n]      = (__bf16)c0[i];
    Yp[r + 16 + n] = (__bf16)c1[i];
    Yp[r + 32 + n] = (__bf16)c2[i];
    Yp[r + 48 + n] = (__bf16)c3[i];
  }
}

// S[(b*H+h)*T + t] = sum_d Xh[b,t,h*64+d]^2
__global__ void __launch_bounds__(256) rowsq_kernel(const __bf16* __restrict__ Xh,
                                                    float* __restrict__ S) {
  int idx = blockIdx.x * 256 + threadIdx.x;  // B*T*H = 65536 threads
  int h = idx & 7, bt = idx >> 3;
  const __bf16* row = Xh + (long)bt * HK + h * KDIM;
  float s = 0.f;
#pragma unroll
  for (int c = 0; c < 8; ++c) {
    v8bf chunk = *(const v8bf*)(row + c * 8);
#pragma unroll
    for (int j = 0; j < 8; ++j) { float f = (float)chunk[j]; s += f * f; }
  }
  int b = bt >> 10, t = bt & 1023;
  S[(((long)(b * HH + h)) << 10) + t] = s;
}

// Fused: S = Q K^T, w = exp(-(q2+k2-2S)), nontemporal store of w (it IS an
// output), ctx += w @ V without re-reading w from HBM. K/V chunks are staged
// into LDS once per block (8 waves share b,h) with double-buffered
// GLOBAL_LOAD_ASYNC_TO_LDS_B128; w tile goes through a wave-private LDS
// transpose (C-layout -> A-layout).
__global__ void __launch_bounds__(256) attn_kernel(const __bf16* __restrict__ Qh,
                                                   const __bf16* __restrict__ Kh,
                                                   const __bf16* __restrict__ Vh,
                                                   const float* __restrict__ Q2,
                                                   const float* __restrict__ K2,
                                                   float* __restrict__ Wout,
                                                   __bf16* __restrict__ Ctx) {
  __shared__ float stage_all[8][16 * LDS_STRIDE];
  __shared__ alignas(32) __bf16 Klds[2][32][KDIM];   // 2 x 4KB
  __shared__ alignas(32) __bf16 Vlds[2][32][KDIM];   // 2 x 4KB
  int tid = threadIdx.x;
  int lane = tid & 31, wave = tid >> 5;
  int bh = blockIdx.x >> 3;                       // b*H + h (shared by block)
  int tm = ((blockIdx.x & 7) << 3) + wave;        // query 16-row block
  int b = bh >> 3, h = bh & 7;
  int half = lane >> 4, nn = lane & 15;

  // cooperative copy mapping: 256 threads x 16B cover one 32x64 bf16 chunk
  int cr = tid >> 3;                              // 0..31 (key row in chunk)
  int cc = (tid & 7) * 8;                         // 0,8,..,56 (head-dim col)
  const __bf16* Kg = Kh + (long)(b * TT) * HK + h * KDIM;
  const __bf16* Vg = Vh + (long)(b * TT) * HK + h * KDIM;

  // preload chunk 0
  async_copy16(Kg + (long)(cr) * HK + cc, &Klds[0][cr][cc]);
  async_copy16(Vg + (long)(cr) * HK + cc, &Vlds[0][cr][cc]);

  const __bf16* Qbase = Qh + (long)(b * TT + tm * 16) * HK + h * KDIM;
  v16bf aq0 = load_A_bf(Qbase, HK, lane);        // d = 0..31
  v16bf aq1 = load_A_bf(Qbase + 32, HK, lane);   // d = 32..63

  float q2r[8];
#pragma unroll
  for (int i = 0; i < 8; ++i) q2r[i] = Q2[((long)bh << 10) + tm * 16 + i + half * 8];

  v8f acc0 = {}, acc1 = {}, acc2 = {}, acc3 = {};
  float* wrow = Wout + (((long)bh << 10) + tm * 16) * (long)TT;
  float* stage = stage_all[wave];

  for (int it = 0; it < TT / 32; ++it) {          // 32 key chunks of 32
    int buf = it & 1;
    if (it + 1 < TT / 32) {                       // prefetch next chunk
      async_copy16(Kg + (long)((it + 1) * 32 + cr) * HK + cc, &Klds[buf ^ 1][cr][cc]);
      async_copy16(Vg + (long)((it + 1) * 32 + cr) * HK + cc, &Vlds[buf ^ 1][cr][cc]);
      asm volatile("s_wait_asynccnt 0x2" ::: "memory");  // older (cur) fill done
    } else {
      asm volatile("s_wait_asynccnt 0x0" ::: "memory");
    }
    __syncthreads();                              // all waves' fills published

#pragma unroll
    for (int sub = 0; sub < 2; ++sub) {
      int tn = it * 2 + sub;
      const __bf16* Kr = &Klds[buf][sub * 16][0]; // rows = keys, cols = d
      v16bf bk0 = load_B_nk_bf(Kr, KDIM, lane);
      v16bf bk1 = load_B_nk_bf(Kr + 32, KDIM, lane);
      v8f s = {};
      s = WMMA_BF16(aq0, bk0, s);
      s = WMMA_BF16(aq1, bk1, s);
      float k2v = K2[((long)bh << 10) + tn * 16 + nn];
#pragma unroll
      for (int i = 0; i < 8; ++i) {
        int m = i + half * 8;
        float w = __expf(-(q2r[i] + k2v - 2.0f * s[i]));   // SIGMA = 1
        __builtin_nontemporal_store(w, wrow + (long)m * TT + tn * 16 + nn);
        stage[m * LDS_STRIDE + sub * 16 + nn] = w;         // C-layout -> LDS
      }
    }
    // wave-private transpose: DS stores complete, then read back in A-layout
    asm volatile("s_wait_dscnt 0x0" ::: "memory");
    v16bf aw = load_A_lds(stage, lane);           // 16x32 w tile as bf16 A
    const __bf16* Vr = &Vlds[buf][0][0];          // rows = keys, cols = d
    v16bf bv0 = load_B_kn_bf(Vr + 0,  KDIM, lane);
    v16bf bv1 = load_B_kn_bf(Vr + 16, KDIM, lane);
    v16bf bv2 = load_B_kn_bf(Vr + 32, KDIM, lane);
    v16bf bv3 = load_B_kn_bf(Vr + 48, KDIM, lane);
    acc0 = WMMA_BF16(aw, bv0, acc0);
    acc1 = WMMA_BF16(aw, bv1, acc1);
    acc2 = WMMA_BF16(aw, bv2, acc2);
    acc3 = WMMA_BF16(aw, bv3, acc3);
    __syncthreads();                              // done reading buf before overwrite
  }

  __bf16* Cp = Ctx + (long)(b * TT + tm * 16) * HK + h * KDIM;
#pragma unroll
  for (int i = 0; i < 8; ++i) {
    int m = i + half * 8;
    Cp[(long)m * HK + nn]      = (__bf16)acc0[i];
    Cp[(long)m * HK + 16 + nn] = (__bf16)acc1[i];
    Cp[(long)m * HK + 32 + nn] = (__bf16)acc2[i];
    Cp[(long)m * HK + 48 + nn] = (__bf16)acc3[i];
  }
}

// out[8192x512](f32) = ctx[8192x512](bf16) @ Wo[512x512](f32); 16x64 strip/wave.
__global__ void __launch_bounds__(256) outproj_kernel(const __bf16* __restrict__ Ctx,
                                                      const float* __restrict__ Wo,
                                                      float* __restrict__ Y) {
  int lane = threadIdx.x & 31, wave = threadIdx.x >> 5;
  int tile = blockIdx.x * 8 + wave;
  int tj = tile & 7;
  int tm = tile >> 3;
  const __bf16* A0 = Ctx + (long)tm * 16 * HK;
  const float* B0 = Wo + tj * 64;
  v8f c0 = {}, c1 = {}, c2 = {}, c3 = {};
  for (int k0 = 0; k0 < HK; k0 += 32) {
    v16bf a = load_A_bf(A0 + k0, HK, lane);
    const float* Bk = B0 + (long)k0 * DD;
    c0 = WMMA_BF16(a, load_B_kn_f32(Bk +  0, DD, lane), c0);
    c1 = WMMA_BF16(a, load_B_kn_f32(Bk + 16, DD, lane), c1);
    c2 = WMMA_BF16(a, load_B_kn_f32(Bk + 32, DD, lane), c2);
    c3 = WMMA_BF16(a, load_B_kn_f32(Bk + 48, DD, lane), c3);
  }
  int half = lane >> 4, n = lane & 15;
  float* Yp = Y + (long)tm * 16 * DD + tj * 64;
#pragma unroll
  for (int i = 0; i < 8; ++i) {
    long r = (long)(i + half * 8) * DD;
    Yp[r + n]      = c0[i];
    Yp[r + 16 + n] = c1[i];
    Yp[r + 32 + n] = c2[i];
    Yp[r + 48 + n] = c3[i];
  }
}

extern "C" void kernel_launch(void* const* d_in, const int* in_sizes, int n_in,
                              void* d_out, int out_size, void* d_ws, size_t ws_size,
                              hipStream_t stream) {
  (void)in_sizes; (void)n_in; (void)out_size; (void)ws_size;
  const float* q  = (const float*)d_in[0];
  const float* k  = (const float*)d_in[1];
  const float* v  = (const float*)d_in[2];
  const float* Wq = (const float*)d_in[3];
  const float* Wk = (const float*)d_in[4];
  const float* Wv = (const float*)d_in[5];
  const float* Wo = (const float*)d_in[6];

  float* out   = (float*)d_out;                       // [B,T,D]
  float* attnw = out + (size_t)BB * TT * DD;          // [B,H,T,T]

  char* w = (char*)d_ws;                              // ~34 MB total
  __bf16* qh  = (__bf16*)w; w += (size_t)BB * TT * HK * sizeof(__bf16);
  __bf16* kh  = (__bf16*)w; w += (size_t)BB * TT * HK * sizeof(__bf16);
  __bf16* vh  = (__bf16*)w; w += (size_t)BB * TT * HK * sizeof(__bf16);
  __bf16* ctx = (__bf16*)w; w += (size_t)BB * TT * HK * sizeof(__bf16);
  float*  q2  = (float*)w;  w += (size_t)BB * HH * TT * sizeof(float);
  float*  k2  = (float*)w;

  dim3 blk(256);
  proj_kernel<<<512, blk, 0, stream>>>(q, Wq, qh);
  proj_kernel<<<512, blk, 0, stream>>>(k, Wk, kh);
  proj_kernel<<<512, blk, 0, stream>>>(v, Wv, vh);
  rowsq_kernel<<<256, blk, 0, stream>>>(qh, q2);
  rowsq_kernel<<<256, blk, 0, stream>>>(kh, k2);
  attn_kernel<<<512, blk, 0, stream>>>(qh, kh, vh, q2, k2, attnw, ctx);
  outproj_kernel<<<512, blk, 0, stream>>>(ctx, Wo, out);
}
